// onering_conv_layer_2190433321287
// MI455X (gfx1250) — matrix operand order repaired and checked by reading, not compiled, not run on hardware
//
#include <hip/hip_runtime.h>
#include <hip/hip_bf16.h>
#include <stdint.h>

// ---------------------------------------------------------------------------
// One-ring spherical conv as a gathered GEMM on CDNA5 (gfx1250).
//   Pipeline per persistent block:
//     [async gather f32 rows -> LDS raw (double buffered, ASYNCcnt)]
//        -> [LDS->LDS convert/transpose f32->bf16]
//        -> [v_wmma_f32_16x16x32_bf16, f32 accum]
//   W^T staged once per block into LDS as bf16.
// ---------------------------------------------------------------------------

typedef __bf16 v16bf __attribute__((ext_vector_type(16)));
typedef float  v8f   __attribute__((ext_vector_type(8)));

#define N_VERT     163842
#define K_NEIGH    7
#define VPT        16                    // vertices per tile (32 (n,b) rows)
#define ROWS_G     (VPT * K_NEIGH)       // 112 gathered vertex rows per tile
#define W_STRIDE   228                   // dwords per W row in LDS (456 bf16)
#define X_STRIDE   68                    // dwords per bf16 row (272B, bank-spread pad)
#define RAW_STRIDE 128                   // dwords per raw f32 row (512B)

static __device__ __forceinline__ unsigned pack_bf16(float a, float b) {
  unsigned ua = __float_as_uint(a), ub = __float_as_uint(b);
  ua = (ua + 0x7FFFu + ((ua >> 16) & 1u)) >> 16;
  ub = (ub + 0x7FFFu + ((ub >> 16) & 1u)) >> 16;
  return ua | (ub << 16);
}

// Low 32 bits of a generic pointer to __shared__ == wave-relative LDS byte
// offset (ISA 10.2).
static __device__ __forceinline__ unsigned lds_off(const void* p) {
  return (unsigned)(unsigned long long)p;
}

static __device__ __forceinline__ void async_gather_b128(unsigned lds_byte,
                                                         const void* gaddr) {
  asm volatile("global_load_async_to_lds_b128 %0, %1, off"
               :: "v"(lds_byte), "v"((unsigned long long)gaddr)
               : "memory");
}

static __device__ __forceinline__ void wait_async_le(void) {
#if __has_builtin(__builtin_amdgcn_s_wait_asynccnt)
  __builtin_amdgcn_s_wait_asynccnt(14);
#else
  asm volatile("s_wait_asynccnt 0xe" ::: "memory");
#endif
}
static __device__ __forceinline__ void wait_async_zero(void) {
#if __has_builtin(__builtin_amdgcn_s_wait_asynccnt)
  __builtin_amdgcn_s_wait_asynccnt(0);
#else
  asm volatile("s_wait_asynccnt 0x0" ::: "memory");
#endif
}

__global__ __launch_bounds__(256)
void onering_conv_wmma(const float* __restrict__ x,
                       const long long* __restrict__ neigh,
                       const float* __restrict__ W,
                       const float* __restrict__ bias,
                       float* __restrict__ out,
                       int nTiles)
{
  __shared__ __align__(16) unsigned w_lds[64 * W_STRIDE];          // 58368 B
  __shared__ __align__(16) unsigned x_raw[2][ROWS_G * RAW_STRIDE]; // 114688 B
  __shared__ __align__(16) unsigned x_bf[ROWS_G * X_STRIDE];       // 30464 B

  const int tid = threadIdx.x;
  const long long idx_count = (long long)K_NEIGH * N_VERT;

  // ---- Stage W (64 x 448 f32) -> LDS bf16, once per persistent block ----
  for (int j = tid; j < 64 * 112; j += 256) {
    const int o = j / 112;
    const int q = j - o * 112;
    const float4 f = *(const float4*)(W + o * 448 + q * 4);
    *(uint2*)(&w_lds[o * W_STRIDE + q * 2]) =
        make_uint2(pack_bf16(f.x, f.y), pack_bf16(f.z, f.w));
  }

  // wave layout: waves 0-3 -> vertices 0-7 of tile, o-tiles 0-3
  //              waves 4-7 -> vertices 8-15,        o-tiles 0-3
  const int wave   = tid >> 5;
  const int lane   = tid & 31;
  const int r      = lane & 15;          // N column of the 16x16 D tile
  const int hi     = lane >> 4;          // lane-half (K / M select)
  const int o_base = (wave & 3) * 16;
  const int vhalf  = (wave >> 2) * 8;
  const int b      = r & 1;
  const int vloc   = vhalf + (r >> 1);

  // ---- issue async gather for this block's first tile (14 issues/wave) ----
  int tile = blockIdx.x;
  const unsigned rawOff0 = lds_off(&x_raw[0][0]);
  const unsigned rawOff1 = lds_off(&x_raw[1][0]);
  if (tile < nTiles) {
    const long long base = (long long)tile * ROWS_G;
    for (int i = tid; i < ROWS_G * 32; i += 256) {
      const int g = i >> 5, q = i & 31;
      const long long gi = base + g;
      int v = 0;
      if (gi < idx_count) v = (int)neigh[gi];
      async_gather_b128(rawOff0 + (unsigned)(g * 512 + q * 16),
                        x + (size_t)v * 128 + q * 4);
    }
  }

  int buf = 0;
  for (; tile < nTiles; tile += gridDim.x) {
    // ---- prefetch next persistent tile into the other raw buffer ----
    const int nextTile = tile + (int)gridDim.x;
    if (nextTile < nTiles) {
      const unsigned nOff = buf ? rawOff0 : rawOff1;
      const long long base = (long long)nextTile * ROWS_G;
      for (int i = tid; i < ROWS_G * 32; i += 256) {
        const int g = i >> 5, q = i & 31;
        const long long gi = base + g;
        int v = 0;
        if (gi < idx_count) v = (int)neigh[gi];
        async_gather_b128(nOff + (unsigned)(g * 512 + q * 16),
                          x + (size_t)v * 128 + q * 4);
      }
      wait_async_le();    // current tile's 14 transfers done; next in flight
    } else {
      wait_async_zero();
    }
    __syncthreads();      // raw[buf] visible to all waves (also covers W stage)

    // ---- LDS->LDS convert/transpose: [c][b] f32 -> [b][c] bf16 ----
    const unsigned* raw = &x_raw[buf][0];
    for (int i = tid; i < ROWS_G * 32; i += 256) {
      const int g = i >> 5, q = i & 31;
      const float4 f = *(const float4*)(raw + g * RAW_STRIDE + q * 4);
      x_bf[g * X_STRIDE + q]      = pack_bf16(f.x, f.z);  // b=0 plane
      x_bf[g * X_STRIDE + 32 + q] = pack_bf16(f.y, f.w);  // b=1 plane
    }
    __syncthreads();

    // ---- 16x16 D tile per wave, K = 448 in 14 steps of 32 ----
    v8f acc = {};
    const unsigned* wrow = &w_lds[(o_base + r) * W_STRIDE + hi * 8];
    const unsigned* xrow = &x_bf[(vloc * K_NEIGH) * X_STRIDE + b * 32 + hi * 8];

#pragma unroll
    for (int k = 0; k < K_NEIGH; ++k) {
#pragma unroll
      for (int h = 0; h < 2; ++h) {
        union { uint4 u[2]; v16bf v; } A, B;
        // A (16-bit 16x32): lanes<16 K 0..15; lanes>=16 K 16..31
        const unsigned* ap = wrow + k * 32 + h * 16;
        A.u[0] = *(const uint4*)(ap);
        A.u[1] = *(const uint4*)(ap + 4);
        // B (16-bit 32x16): lanes<16 K 0..15; lanes>=16 K 16..31
        const unsigned* bp = xrow + k * X_STRIDE + h * 16;
        B.u[0] = *(const uint4*)(bp);
        B.u[1] = *(const uint4*)(bp + 4);
        acc = __builtin_amdgcn_wmma_f32_16x16x32_bf16(
            false, A.v, false, B.v, (short)0, acc, false, false);
      }
    }

    // ---- epilogue: bias + store (D layout: VGPR v -> o = o_base + v + hi*8) ----
    const int n = tile * VPT + vloc;
    if (n < N_VERT) {
      const float* bv = bias + o_base + hi * 8;
      float* op = out + (size_t)n * 128 + (o_base + hi * 8) * 2 + b;
#pragma unroll
      for (int v = 0; v < 8; ++v) op[v * 2] = acc[v] + bv[v];
    }
    __syncthreads();      // x_bf reuse + raw[buf] will be re-filled next iter
    buf ^= 1;
  }
}

extern "C" void kernel_launch(void* const* d_in, const int* in_sizes, int n_in,
                              void* d_out, int out_size, void* d_ws, size_t ws_size,
                              hipStream_t stream) {
  const float*     x     = (const float*)d_in[0];
  const long long* neigh = (const long long*)d_in[1];
  const float*     W     = (const float*)d_in[2];
  const float*     bias  = (const float*)d_in[3];
  float*           out   = (float*)d_out;

  const int nTiles = (N_VERT + VPT - 1) / VPT;   // 10241
  const int grid   = nTiles < 2048 ? nTiles : 2048;
  onering_conv_wmma<<<grid, 256, 0, stream>>>(x, neigh, W, bias, out, nTiles);
}
